// VQ_82729660056080
// MI455X (gfx1250) — matrix operand order repaired and checked by reading, not compiled, not run on hardware
//
#include <hip/hip_runtime.h>

typedef __attribute__((ext_vector_type(16))) _Float16 v16h;
typedef __attribute__((ext_vector_type(8)))  float    v8f;

#define K_CODES 512
#define DIM     64
#define HW      4096                  // 64*64
#define N_IMG   32
#define NROWS   (N_IMG * HW)          // 131072
#define ROWS_PB 128
#define NBLOCKS (NROWS / ROWS_PB)     // 1024
#define THREADS 256
#define BETA    0.25f

// workspace layout (bytes)
#define WS_CB_OFF   0                     // K_CODES*DIM f16  = 65536 B
#define WS_E2_OFF   65536                 // K_CODES  f32     =  2048 B
#define WS_PART_OFF (65536 + 2048)        // NBLOCKS f32      =  4096 B

// ---------------------------------------------------------------------------
// Prep: fp32 codebook -> f16 workspace copy; per-code squared norms e2[k].
// ---------------------------------------------------------------------------
__global__ void vq_prep(const float* __restrict__ emb,
                        _Float16* __restrict__ cb16,
                        float* __restrict__ e2) {
    int gid = blockIdx.x * blockDim.x + threadIdx.x;
    if (gid < K_CODES * DIM) cb16[gid] = (_Float16)emb[gid];
    if (gid < K_CODES) {
        const float* row = emb + gid * DIM;
        float s = 0.f;
        #pragma unroll 8
        for (int d = 0; d < DIM; ++d) { float v = row[d]; s += v * v; }
        e2[gid] = s;
    }
}

// ---------------------------------------------------------------------------
// Main: per block 128 rows. WMMA f16 (fp32 acc) distances, argmin, gather,
// NCHW write-back, loss partial.
// ---------------------------------------------------------------------------
__global__ __launch_bounds__(THREADS)
void vq_main(const float* __restrict__ x,
             const float* __restrict__ emb,
             const _Float16* __restrict__ cb16,
             const float* __restrict__ e2g,
             float* __restrict__ out,
             float* __restrict__ partials) {
    __shared__ _Float16 sE[K_CODES][DIM + 8];   // 73728 B, row stride rotates 4 banks
    __shared__ float    sX[ROWS_PB][DIM + 1];   // 33280 B, row stride rotates 1 bank
    __shared__ float    sE2[K_CODES];           //  2048 B
    __shared__ int      sIdx[ROWS_PB];          //   512 B
    __shared__ float    sRed[THREADS / 32];

    const int tid  = threadIdx.x;
    const int lane = tid & 31;
    const int wave = tid >> 5;

    // stage f16 codebook (dword-wide) + e2 into LDS (coalesced)
    {
        const uint32_t* cb32 = (const uint32_t*)cb16;
        for (int i = tid; i < K_CODES * DIM / 2; i += THREADS) {
            int code = i >> 5, d2 = (i & 31) << 1;          // 32 dwords per code row
            *(uint32_t*)&sE[code][d2] = cb32[i];
        }
        for (int i = tid; i < K_CODES; i += THREADS) sE2[i] = e2g[i];
    }

    // stage X tile: NCHW -> LDS [row][d]  (float4 global loads, coalesced over hw)
    const int n   = blockIdx.x >> 5;               // image
    const int hw0 = (blockIdx.x & 31) * ROWS_PB;   // hw offset
    const float* xb = x + (size_t)n * DIM * HW + hw0;
    for (int i = tid; i < ROWS_PB * DIM / 4; i += THREADS) {
        int d = i >> 5, r4 = (i & 31) << 2;
        float4 v = *(const float4*)&xb[(size_t)d * HW + r4];
        sX[r4 + 0][d] = v.x;
        sX[r4 + 1][d] = v.y;
        sX[r4 + 2][d] = v.z;
        sX[r4 + 3][d] = v.w;
    }
    __syncthreads();

    // ---- per-wave: one 16-row M-tile, argmin over 512 codes ----
    const int m_base = wave * 16;
    const int mrow   = m_base + (lane & 15);
    const int ak     = (lane < 16) ? 0 : 16;    // A K-offset, contiguous k = 16*(lane/16) + s
    v16h a0, a1;
    #pragma unroll
    for (int j = 0; j < 8; ++j) {
        a0[j]     = (_Float16)sX[mrow][ak + j];
        a0[j + 8] = (_Float16)sX[mrow][ak + 8 + j];
        a1[j]     = (_Float16)sX[mrow][32 + ak + j];
        a1[j + 8] = (_Float16)sX[mrow][32 + ak + 8 + j];
    }
    const int bk = (lane < 16) ? 0 : 16;        // B K-offset, contiguous k = 16*(lane/16) + s

    float best[8]; int bidx[8];
    #pragma unroll
    for (int s = 0; s < 8; ++s) { best[s] = 3.4e38f; bidx[s] = 0; }

    // Within a lane the visited code index strictly increases, so strict '<'
    // keeps the first (lowest) index on ties; cross-lane ties are resolved in
    // the shuffle reduction below.
    #pragma unroll 2
    for (int nt = 0; nt < K_CODES / 16; ++nt) {
        const int col = nt * 16 + (lane & 15);  // code index this lane scores
        v16h b0, b1;
        #pragma unroll
        for (int j = 0; j < 16; ++j) {
            b0[j] = sE[col][bk + j];
            b1[j] = sE[col][32 + bk + j];
        }
        v8f c = {};
        c = __builtin_amdgcn_wmma_f32_16x16x32_f16(false, a0, false, b0,
                                                   (short)0, c, false, false);
        c = __builtin_amdgcn_wmma_f32_16x16x32_f16(false, a1, false, b1,
                                                   (short)0, c, false, false);
        const float e2n = sE2[col];
        #pragma unroll
        for (int s = 0; s < 8; ++s) {           // dist ~ e2 - 2*(x.e)  (x2 const/row)
            float sc = __builtin_fmaf(-2.0f, c[s], e2n);
            if (sc < best[s]) { best[s] = sc; bidx[s] = col; }
        }
    }

    // argmin across the 16 lanes sharing each row (lanes 0-15: M=s, 16-31: M=s+8)
    #pragma unroll
    for (int off = 1; off < 16; off <<= 1) {
        #pragma unroll
        for (int s = 0; s < 8; ++s) {
            float osc = __shfl_xor(best[s], off, 32);
            int   oix = __shfl_xor(bidx[s], off, 32);
            if (osc < best[s] || (osc == best[s] && oix < bidx[s])) {
                best[s] = osc; bidx[s] = oix;
            }
        }
    }
    if ((lane & 15) == 0) {
        int mo = m_base + ((lane >> 4) ? 8 : 0);
        #pragma unroll
        for (int s = 0; s < 8; ++s) sIdx[mo + s] = bidx[s];
    }
    __syncthreads();

    // ---- gather fp32 code rows, write NCHW (float4), accumulate loss ----
    float lsum = 0.f;
    float* ob = out + (size_t)n * DIM * HW + hw0;
    for (int i = tid; i < ROWS_PB * DIM / 4; i += THREADS) {
        int d = i >> 5, r4 = (i & 31) << 2;
        float4 q;
        q.x = emb[sIdx[r4 + 0] * DIM + d];      // exact fp32 codebook (L2 resident)
        q.y = emb[sIdx[r4 + 1] * DIM + d];
        q.z = emb[sIdx[r4 + 2] * DIM + d];
        q.w = emb[sIdx[r4 + 3] * DIM + d];
        float d0 = q.x - sX[r4 + 0][d];
        float d1 = q.y - sX[r4 + 1][d];
        float d2 = q.z - sX[r4 + 2][d];
        float d3 = q.w - sX[r4 + 3][d];
        lsum += d0 * d0 + d1 * d1 + d2 * d2 + d3 * d3;
        *(float4*)&ob[(size_t)d * HW + r4] = q; // quantized_st == quantized (value)
    }
    #pragma unroll
    for (int off = 16; off >= 1; off >>= 1) lsum += __shfl_xor(lsum, off, 32);
    if (lane == 0) sRed[wave] = lsum;
    __syncthreads();
    if (tid == 0) {
        float t = 0.f;
        #pragma unroll
        for (int w = 0; w < THREADS / 32; ++w) t += sRed[w];
        partials[blockIdx.x] = t;
    }
}

// ---------------------------------------------------------------------------
// Deterministic fixed-order loss reduction: loss = (1+beta)*mean((q-x)^2)
// ---------------------------------------------------------------------------
__global__ void vq_loss_reduce(const float* __restrict__ partials,
                               float* __restrict__ loss_out) {
    __shared__ float sb[THREADS / 32];
    float s = 0.f;
    for (int i = threadIdx.x; i < NBLOCKS; i += THREADS) s += partials[i];
    #pragma unroll
    for (int off = 16; off >= 1; off >>= 1) s += __shfl_xor(s, off, 32);
    const int lane = threadIdx.x & 31, wave = threadIdx.x >> 5;
    if (lane == 0) sb[wave] = s;
    __syncthreads();
    if (threadIdx.x == 0) {
        float t = 0.f;
        #pragma unroll
        for (int w = 0; w < THREADS / 32; ++w) t += sb[w];
        loss_out[0] = (1.0f + BETA) * t / (float)((size_t)NROWS * DIM);
    }
}

extern "C" void kernel_launch(void* const* d_in, const int* in_sizes, int n_in,
                              void* d_out, int out_size, void* d_ws, size_t ws_size,
                              hipStream_t stream) {
    (void)in_sizes; (void)n_in; (void)ws_size;
    const float* x   = (const float*)d_in[0];   // [32,64,64,64] NCHW fp32
    const float* emb = (const float*)d_in[1];   // [512,64] fp32

    float* out = (float*)d_out;                 // 8388608 quantized + 1 loss
    char*  ws  = (char*)d_ws;
    _Float16* cb16  = (_Float16*)(ws + WS_CB_OFF);
    float*    e2    = (float*)(ws + WS_E2_OFF);
    float*    parts = (float*)(ws + WS_PART_OFF);

    vq_prep<<<(K_CODES * DIM + THREADS - 1) / THREADS, THREADS, 0, stream>>>(emb, cb16, e2);
    vq_main<<<NBLOCKS, THREADS, 0, stream>>>(x, emb, cb16, e2, out, parts);
    vq_loss_reduce<<<1, THREADS, 0, stream>>>(parts, out + (out_size - 1));
}
